// Model_81956565942989
// MI455X (gfx1250) — compile-verified, matrix operand order checked by reading, not run on hardware
//
#include <hip/hip_runtime.h>
#include <hip/hip_bf16.h>
#include <math.h>

// ---------------------------------------------------------------------------
// CDNA5 (gfx1250) wave32 WMMA bf16 pipeline for crop->LSTM(x3)->attention.
// All heavy math goes through v_wmma_f32_16x16x32_bf16 with f32 accumulation.
// Weights are converted f32->bf16 on the fly inside the GEMM (bandwidth-
// optimal: each weight byte is read from HBM exactly once per use, no extra
// conversion pass traffic). Activations are staged as bf16.
// ---------------------------------------------------------------------------

typedef __attribute__((ext_vector_type(16))) __bf16 v16bf;
typedef __attribute__((ext_vector_type(8)))  float  v8f;

#define NB 64
#define PP 7

__device__ __forceinline__ unsigned short f2bf(float f) {
  unsigned int u = __float_as_uint(f);
  unsigned int r = (u + 0x7FFFu + ((u >> 16) & 1u)) >> 16;  // round-nearest-even
  return (unsigned short)r;
}
__device__ __forceinline__ float bf2f(unsigned short u) {
  return __uint_as_float(((unsigned int)u) << 16);
}
__device__ __forceinline__ float sigm(float x) { return 1.0f / (1.0f + expf(-x)); }

union TileAB {
  v16bf v;
  uint4 q[2];
  __bf16 h[16];
  unsigned short u[16];
};

// ---------------------------------------------------------------------------
// Generic GEMM: C[M,Nout] = act(A[M,K](bf16) * W[Nout,K](f32)^T + bias)
// One wave (32 threads) per block computes a 64x64 output tile via 16 WMMAs
// per K-step of 32. A-operand VGPR layout per ISA 7.12.2 (16-bit A 16x32):
//   lane l: m = l&15, hi = l>>4; holds K = {hi*8..hi*8+7} and {hi*8+16..+23}
// B-operand: lane n holds column n of B (= row n of W), same K chunking.
// Requires: M % 64 == 0, K % 32 == 0.
// Nout handling: weight-row index is CLAMPED (not branched) — B lane n only
// feeds D column n, and columns >= Nout are never stored, so out-of-range
// lanes may load any valid row. This keeps the hot loop branch-free.
// ---------------------------------------------------------------------------
__global__ __launch_bounds__(32)
void gemm_bf16_wmma(const unsigned short* __restrict__ A, int lda,
                    const float* __restrict__ W, int K, int Nout,
                    const float* __restrict__ bias,
                    float* __restrict__ Cf, unsigned short* __restrict__ Cb,
                    int ldc, int relu)
{
  const int lane = threadIdx.x;
  const int nlo  = lane & 15;
  const int hi   = lane >> 4;
  const int mBase = blockIdx.x * 64;
  const int nBase = blockIdx.y * 64;

  // Clamped weight rows for the 4 n-subtiles (always in-range loads).
  const float* wrow[4];
#pragma unroll
  for (int j = 0; j < 4; ++j) {
    int nrow = nBase + j * 16 + nlo;
    if (nrow > Nout - 1) nrow = Nout - 1;
    wrow[j] = W + (size_t)nrow * K;
  }

  v8f acc[4][4] = {};

  for (int k0 = 0; k0 < K; k0 += 32) {
    TileAB a[4];
#pragma unroll
    for (int i = 0; i < 4; ++i) {
      const unsigned short* ap =
          A + (size_t)(mBase + i * 16 + nlo) * lda + k0 + hi * 8;
      a[i].q[0] = *(const uint4*)ap;
      a[i].q[1] = *(const uint4*)(ap + 16);
    }
#pragma unroll
    for (int j = 0; j < 4; ++j) {
      TileAB b;
      const float* wp = wrow[j] + k0 + hi * 8;
      float4 w0 = ((const float4*)wp)[0];
      float4 w1 = ((const float4*)wp)[1];
      float4 w2 = ((const float4*)(wp + 16))[0];
      float4 w3 = ((const float4*)(wp + 16))[1];
      b.h[0]  = (__bf16)w0.x; b.h[1]  = (__bf16)w0.y; b.h[2]  = (__bf16)w0.z; b.h[3]  = (__bf16)w0.w;
      b.h[4]  = (__bf16)w1.x; b.h[5]  = (__bf16)w1.y; b.h[6]  = (__bf16)w1.z; b.h[7]  = (__bf16)w1.w;
      b.h[8]  = (__bf16)w2.x; b.h[9]  = (__bf16)w2.y; b.h[10] = (__bf16)w2.z; b.h[11] = (__bf16)w2.w;
      b.h[12] = (__bf16)w3.x; b.h[13] = (__bf16)w3.y; b.h[14] = (__bf16)w3.z; b.h[15] = (__bf16)w3.w;
#pragma unroll
      for (int i = 0; i < 4; ++i) {
        acc[i][j] = __builtin_amdgcn_wmma_f32_16x16x32_bf16(
            false, a[i].v, false, b.v, (short)0, acc[i][j], false, false);
      }
    }
  }

  // Epilogue: C/D layout per ISA 7.12.2 (32-bit 16x16):
  //   lane l: N = l&15; VGPR r holds M = 8*(l>>4) + r.
#pragma unroll
  for (int j = 0; j < 4; ++j) {
    const int col = nBase + j * 16 + nlo;
    if (col >= Nout) continue;
    const float bv = bias ? bias[col] : 0.0f;
#pragma unroll
    for (int i = 0; i < 4; ++i) {
#pragma unroll
      for (int r = 0; r < 8; ++r) {
        const int row = mBase + i * 16 + hi * 8 + r;
        float v = acc[i][j][r] + bv;
        if (relu) v = v > 0.0f ? v : 0.0f;
        if (Cf) Cf[(size_t)row * ldc + col] = v;
        if (Cb) Cb[(size_t)row * ldc + col] = f2bf(v);
      }
    }
  }
}

// ---------------------------------------------------------------------------
// Box average pooling (direct summation; cheaper than integral image in HBM
// bytes). One block per (n,p); threads stride over channels. Writes bf16
// directly into the (possibly concatenated) GEMM A-buffer.
// ---------------------------------------------------------------------------
__global__ void crop_avg_kernel(const float* __restrict__ x,
                                const int* __restrict__ patchs,
                                unsigned short* __restrict__ A,
                                int C, int H, int num, int den,
                                int ldA, int colOff)
{
  const int np = blockIdx.x;  // 0..447
  const int n = np / PP;
  const int* pb = patchs + (size_t)np * 4;
  const int p0 = pb[0], p1 = pb[1], p2 = pb[2], p3 = pb[3];
  const bool zero = ((p0 | p1 | p2 | p3) == 0);
  const int y0 = (p0 * num) / den, y1 = (p1 * num) / den;
  const int x0 = (p2 * num) / den, x1 = (p3 * num) / den;
  const float inv = 1.0f / (float)((y1 - y0 + 1) * (x1 - x0 + 1));

  for (int c = threadIdx.x; c < C; c += blockDim.x) {
    float s = 0.0f;
    if (!zero) {
      const float* base = x + ((size_t)n * C + c) * H * H;
      for (int y = y0; y <= y1; ++y) {
        const float* row = base + (size_t)y * H;
        for (int xx = x0; xx <= x1; ++xx) s += row[xx];
      }
      s *= inv;
    }
    A[(size_t)np * ldA + colOff + c] = f2bf(s);
  }
}

// ---------------------------------------------------------------------------
// LSTM cell (torch gate order i,f,g,o). Reads precomputed input gates xg and
// (t>0) recurrent gates hg, updates c, writes h as bf16 both to the
// recurrence buffer and into the next layer's concatenated A-buffer.
// ---------------------------------------------------------------------------
__global__ void lstm_cell_kernel(const float* __restrict__ xg,
                                 const float* __restrict__ hg,
                                 float* __restrict__ c,
                                 unsigned short* __restrict__ hbuf,
                                 unsigned short* __restrict__ dest,
                                 int H, int t, int sN, int sT, int off)
{
  const int id = blockIdx.x * blockDim.x + threadIdx.x;
  if (id >= NB * H) return;
  const int n = id / H, j = id % H;
  const size_t xrow = (size_t)(n * PP + t) * 4 * H;
  float gi = xg[xrow + j];
  float gf = xg[xrow + H + j];
  float gg = xg[xrow + 2 * H + j];
  float go = xg[xrow + 3 * H + j];
  if (hg) {
    const size_t hrow = (size_t)n * 4 * H;
    gi += hg[hrow + j];
    gf += hg[hrow + H + j];
    gg += hg[hrow + 2 * H + j];
    go += hg[hrow + 3 * H + j];
  }
  const float cp = t ? c[(size_t)n * H + j] : 0.0f;
  const float cn = sigm(gf) * cp + sigm(gi) * tanhf(gg);
  const float hn = sigm(go) * tanhf(cn);
  c[(size_t)n * H + j] = cn;
  const unsigned short hb = f2bf(hn);
  hbuf[(size_t)n * H + j] = hb;
  dest[(size_t)n * sN + (size_t)t * sT + off + j] = hb;
}

// ---------------------------------------------------------------------------
// Softmax over the 7 patch scores per batch row.
// ---------------------------------------------------------------------------
__global__ void softmax7_kernel(const float* __restrict__ attn,
                                float* __restrict__ alpha)
{
  const int n = blockIdx.x * blockDim.x + threadIdx.x;
  if (n >= NB) return;
  float v[PP];
  float m = -3.4e38f;
#pragma unroll
  for (int p = 0; p < PP; ++p) { v[p] = attn[n * PP + p]; m = v[p] > m ? v[p] : m; }
  float s = 0.0f;
#pragma unroll
  for (int p = 0; p < PP; ++p) { v[p] = expf(v[p] - m); s += v[p]; }
  const float inv = 1.0f / s;
#pragma unroll
  for (int p = 0; p < PP; ++p) alpha[n * PP + p] = v[p] * inv;
}

// ---------------------------------------------------------------------------
// feat[n,j] = sum_t rnn3[n,t,j] * alpha[n,t]   (bf16 out, feeds final GEMM)
// ---------------------------------------------------------------------------
__global__ void wsum_kernel(const unsigned short* __restrict__ rnn3,
                            const float* __restrict__ alpha,
                            unsigned short* __restrict__ feat)
{
  const int id = blockIdx.x * blockDim.x + threadIdx.x;
  if (id >= NB * 1792) return;
  const int n = id / 1792, j = id % 1792;
  float s = 0.0f;
#pragma unroll
  for (int t = 0; t < PP; ++t)
    s += bf2f(rnn3[(size_t)n * 12544 + (size_t)t * 1792 + j]) * alpha[n * PP + t];
  feat[(size_t)n * 1792 + j] = f2bf(s);
}

// ---------------------------------------------------------------------------
extern "C" void kernel_launch(void* const* d_in, const int* in_sizes, int n_in,
                              void* d_out, int out_size, void* d_ws, size_t ws_size,
                              hipStream_t stream) {
  (void)in_sizes; (void)n_in; (void)out_size; (void)ws_size;

  const float* x1     = (const float*)d_in[0];
  const float* x2     = (const float*)d_in[1];
  const float* x3     = (const float*)d_in[2];
  const int*   patchs = (const int*)  d_in[3];
  const float* l1_wih = (const float*)d_in[4];
  const float* l1_whh = (const float*)d_in[5];
  const float* l1_b   = (const float*)d_in[6];
  const float* l2_wih = (const float*)d_in[7];
  const float* l2_whh = (const float*)d_in[8];
  const float* l2_b   = (const float*)d_in[9];
  const float* l3_wih = (const float*)d_in[10];
  const float* l3_whh = (const float*)d_in[11];
  const float* l3_b   = (const float*)d_in[12];
  const float* aw1    = (const float*)d_in[13];
  const float* ab1    = (const float*)d_in[14];
  const float* aw2    = (const float*)d_in[15];
  const float* ab2    = (const float*)d_in[16];
  const float* aw3    = (const float*)d_in[17];
  const float* ab3    = (const float*)d_in[18];
  const float* fw     = (const float*)d_in[19];
  const float* fb     = (const float*)d_in[20];

  // Bump allocator over workspace (256B aligned).
  char* ws = (char*)d_ws;
  size_t off = 0;
  auto alloc = [&](size_t bytes) -> void* {
    void* p = ws + off;
    off = (off + bytes + 255) & ~(size_t)255;
    return p;
  };

  const int M = NB * PP;  // 448
  unsigned short* A1   = (unsigned short*)alloc((size_t)M * 256 * 2);
  unsigned short* A2   = (unsigned short*)alloc((size_t)M * 768 * 2);
  unsigned short* A3   = (unsigned short*)alloc((size_t)M * 1792 * 2);
  unsigned short* flat = (unsigned short*)alloc((size_t)NB * 12544 * 2);  // rnn3
  float*          xg   = (float*)alloc((size_t)M * 7168 * 4);   // reused per layer
  float*          hg   = (float*)alloc((size_t)NB * 7168 * 4);  // reused per layer
  unsigned short* hbuf = (unsigned short*)alloc((size_t)NB * 1792 * 2);
  float*          cbuf = (float*)alloc((size_t)NB * 1792 * 4);
  unsigned short* h1a  = (unsigned short*)alloc((size_t)NB * 3584 * 2);
  unsigned short* h2a  = (unsigned short*)alloc((size_t)NB * 1792 * 2);
  float*          attn = (float*)alloc((size_t)NB * PP * 4);
  float*          alph = (float*)alloc((size_t)NB * PP * 4);
  unsigned short* feat = (unsigned short*)alloc((size_t)NB * 1792 * 2);

  // ---- Crop + average pool into concatenated A-buffers (bf16) ----
  crop_avg_kernel<<<M, 256, 0, stream>>>(x1, patchs, A1, 256, 56, 55, 55, 256, 0);
  crop_avg_kernel<<<M, 256, 0, stream>>>(x2, patchs, A2, 512, 28, 27, 55, 768, 0);
  crop_avg_kernel<<<M, 256, 0, stream>>>(x3, patchs, A3, 1024, 14, 13, 55, 1792, 0);

  // ---- LSTM layers ----
  struct L {
    const unsigned short* A; int D; const float *wih, *whh, *b; int H;
    unsigned short* dest; int sN, sT, offc;
  } layers[3] = {
    { A1, 256,  l1_wih, l1_whh, l1_b, 256,  A2,   PP * 768,  768,  512  },
    { A2, 768,  l2_wih, l2_whh, l2_b, 768,  A3,   PP * 1792, 1792, 1024 },
    { A3, 1792, l3_wih, l3_whh, l3_b, 1792, flat, 12544,     1792, 0    },
  };
  for (int li = 0; li < 3; ++li) {
    const L& L_ = layers[li];
    const int G = 4 * L_.H;
    // xg = A @ wih^T + b   (M=448)
    gemm_bf16_wmma<<<dim3(M / 64, G / 64), 32, 0, stream>>>(
        L_.A, L_.D, L_.wih, L_.D, G, L_.b, xg, nullptr, G, 0);
    for (int t = 0; t < PP; ++t) {
      const float* hgp = nullptr;
      if (t) {
        gemm_bf16_wmma<<<dim3(1, G / 64), 32, 0, stream>>>(
            hbuf, L_.H, L_.whh, L_.H, G, nullptr, hg, nullptr, G, 0);
        hgp = hg;
      }
      lstm_cell_kernel<<<(NB * L_.H + 255) / 256, 256, 0, stream>>>(
          xg, hgp, cbuf, hbuf, L_.dest, L_.H, t, L_.sN, L_.sT, L_.offc);
    }
  }

  // ---- Attention MLP ----
  gemm_bf16_wmma<<<dim3(1, 3584 / 64), 32, 0, stream>>>(
      flat, 12544, aw1, 12544, 3584, ab1, nullptr, h1a, 3584, 1);
  gemm_bf16_wmma<<<dim3(1, 1792 / 64), 32, 0, stream>>>(
      h1a, 3584, aw2, 3584, 1792, ab2, nullptr, h2a, 1792, 1);
  gemm_bf16_wmma<<<dim3(1, 1), 32, 0, stream>>>(
      h2a, 1792, aw3, 1792, PP, ab3, attn, nullptr, PP, 1);
  softmax7_kernel<<<1, 64, 0, stream>>>(attn, alph);
  wsum_kernel<<<(NB * 1792 + 255) / 256, 256, 0, stream>>>(flat, alph, feat);

  // ---- Final projection -> d_out [64,512] f32 ----
  gemm_bf16_wmma<<<dim3(1, 512 / 64), 32, 0, stream>>>(
      feat, 1792, fw, 1792, 512, fb, (float*)d_out, nullptr, 512, 0);
}
